// BinaryAssociativeMemory_24696061952675
// MI455X (gfx1250) — matrix operand order repaired and checked by reading, not compiled, not run on hardware
//
#include <hip/hip_runtime.h>
#include <hip/hip_bf16.h>

typedef __attribute__((ext_vector_type(16))) _Float16 v16h;
typedef __attribute__((ext_vector_type(8)))  float    v8f;
typedef unsigned int uint;
typedef int v4i __attribute__((vector_size(16)));   // matches builtin's pointee type

// ---------- constants for this problem ----------
// B=2, T=4096, D=2048, H=16, dh=128, C=64, N=T/C=64, BT=8192
#define BT_   8192
#define D_    2048
#define T_    4096
#define H_    16
#define DH_   128
#define CC_   64

#if __has_builtin(__builtin_amdgcn_global_load_async_to_lds_b128)
#define HAS_ASYNC_LDS 1
#else
#define HAS_ASYNC_LDS 0
#endif

union FragU {
    v16h v;
    uint4 q[2];
    _Float16 h[16];
};

__device__ inline v8f zeroV() {
    v8f z = {0.f,0.f,0.f,0.f,0.f,0.f,0.f,0.f};
    return z;
}

__device__ inline v8f wmma32(v16h a, v16h b, v8f c) {
    // (neg_a, A, neg_b, B, c_mod, C, reuse_a, reuse_b)
    return __builtin_amdgcn_wmma_f32_16x16x32_f16(false, a, false, b, (short)0, c, false, false);
}

// ---- async global->LDS (gfx1250 GLOBAL_LOAD_ASYNC_TO_LDS_B128, ASYNCcnt) ----
typedef __attribute__((address_space(1))) v4i* as1_v4i_ptr;
typedef __attribute__((address_space(3))) v4i* as3_v4i_ptr;

__device__ inline void copy_b128(const _Float16* g, _Float16* l) {
#if HAS_ASYNC_LDS
    // integer round-trip: always-legal AS casts; low 32 bits of a generic LDS
    // address are the LDS offset per the ISA's aperture mapping.
    __builtin_amdgcn_global_load_async_to_lds_b128(
        (as1_v4i_ptr)(size_t)g, (as3_v4i_ptr)(size_t)l, 0, 0);
#else
    *(uint4*)l = *(const uint4*)g;
#endif
}

template <int N>
__device__ inline void async_wait() {
#if HAS_ASYNC_LDS
#if __has_builtin(__builtin_amdgcn_s_wait_asynccnt)
    __builtin_amdgcn_s_wait_asynccnt((unsigned short)N);
#else
    asm volatile("s_wait_asynccnt %0" :: "i"(N) : "memory");
#endif
#endif
}

// A operand: row-major MxK, lane(0-15)=row, 16 contiguous K at kBase+(lane>>4)*16
__device__ inline v16h load_fragA(const _Float16* S, int stride, int mBase, int kBase, int lane) {
    int lr = lane & 15, lh = lane >> 4;
    const _Float16* p = S + (mBase + lr) * stride + kBase + lh * 16;
    FragU f;
    f.q[0] = *(const uint4*)p;
    f.q[1] = *(const uint4*)(p + 8);
    return f.v;
}

// B operand from Bt stored N x K row-major: lane(0-15)=col n, 16 contiguous K at kBase+(lane>>4)*16
__device__ inline v16h load_fragB(const _Float16* S, int stride, int nBase, int kBase, int lane) {
    int lr = lane & 15, lh = lane >> 4;
    const _Float16* p = S + (nBase + lr) * stride + kBase + lh * 16;
    FragU f;
    f.q[0] = *(const uint4*)p;
    f.q[1] = *(const uint4*)(p + 8);
    return f.v;
}

// B operand from an f32 LDS matrix stored N x K row-major (convert to f16 on the fly)
__device__ inline v16h load_fragB_f32(const float* S, int stride, int nBase, int kBase, int lane) {
    int lr = lane & 15, lh = lane >> 4;
    const float* p = S + (nBase + lr) * stride + kBase + lh * 16;
    FragU f;
#pragma unroll
    for (int i = 0; i < 16; ++i) f.h[i] = (_Float16)p[i];
    return f.v;
}

// ---------------- f32 -> f16 convert ----------------
__global__ void cvt_f32_f16(const float* __restrict__ src, _Float16* __restrict__ dst, int n) {
    int i = blockIdx.x * blockDim.x + threadIdx.x;
    if (i < n) dst[i] = (_Float16)src[i];
}

// ---------------- tiled WMMA GEMM (double-buffered, async LDS fill) ----------------
// C(MxN) = A(MxK, f16 row-major) * B, with Bt (NxK, f16 row-major) supplied.
// mode 0: q epilogue  -> f16, scaled by 1/sqrt(128), scattered to (B,H,T,dh)
// mode 1: sign epilogue -> f16 (+1/-1), scattered to (B,H,T,dh)
// mode 2: plain f32 row-major MxN into Cout
__global__ __launch_bounds__(256) void gemm_f16(
    const _Float16* __restrict__ A, const _Float16* __restrict__ Bt,
    void* __restrict__ Cout, int M, int N, int K, int mode)
{
    __shared__ __align__(16) _Float16 As[2][128 * 64];   // 2 x 16 KB
    __shared__ __align__(16) _Float16 Bs[2][128 * 64];   // 2 x 16 KB

    const int tid  = threadIdx.x;
    const int lane = tid & 31;
    const int wave = tid >> 5;
    const int wm = wave >> 2;      // 0..1  -> 64 rows each
    const int wn = wave & 3;       // 0..3  -> 32 cols each

    const int nTilesN = N >> 7;
    const int tileM = (int)(blockIdx.x / nTilesN) << 7;
    const int tileN = (int)(blockIdx.x % nTilesN) << 7;

    v8f acc[4][2];
#pragma unroll
    for (int s = 0; s < 4; ++s)
#pragma unroll
        for (int t = 0; t < 2; ++t) acc[s][t] = zeroV();

    // panel fill: 128 rows x 64 cols (f16) = 1024 b128 segments; 4 per thread per matrix
    auto issue = [&](int kb, int buf) {
#pragma unroll
        for (int i = 0; i < 4; ++i) {
            const int s2  = tid + 256 * i;       // 0..1023
            const int row = s2 >> 3;
            const int off = (s2 & 7) << 3;       // elements
            copy_b128(A  + (size_t)(tileM + row) * K + kb + off, &As[buf][row * 64 + off]);
            copy_b128(Bt + (size_t)(tileN + row) * K + kb + off, &Bs[buf][row * 64 + off]);
        }
    };

    issue(0, 0);
    int cur = 0;
    for (int kb = 0; kb < K; kb += 64) {
        if (kb + 64 < K) { issue(kb + 64, cur ^ 1); async_wait<8>(); }
        else             { async_wait<0>(); }
        __syncthreads();

        const _Float16* Ap = &As[cur][0];
        const _Float16* Bp = &Bs[cur][0];
#pragma unroll
        for (int kk = 0; kk < 64; kk += 32) {
            v16h af[4], bf[2];
#pragma unroll
            for (int s = 0; s < 4; ++s) af[s] = load_fragA(Ap, 64, wm * 64 + s * 16, kk, lane);
#pragma unroll
            for (int t = 0; t < 2; ++t) bf[t] = load_fragB(Bp, 64, wn * 32 + t * 16, kk, lane);
#pragma unroll
            for (int s = 0; s < 4; ++s)
#pragma unroll
                for (int t = 0; t < 2; ++t) acc[s][t] = wmma32(af[s], bf[t], acc[s][t]);
        }
        __syncthreads();   // protects buffer 'cur' before it is refilled next iteration
        cur ^= 1;
    }

    const int lr = lane & 15, lh = lane >> 4;
    const float qscale = 0.08838834764831845f;   // 1/sqrt(128)
#pragma unroll
    for (int s = 0; s < 4; ++s) {
#pragma unroll
        for (int t = 0; t < 2; ++t) {
            const int rowBase = tileM + wm * 64 + s * 16 + lh * 8;
            const int col     = tileN + wn * 32 + t * 16 + lr;
#pragma unroll
            for (int r = 0; r < 8; ++r) {
                const int row = rowBase + r;
                const float val = acc[s][t][r];
                if (mode == 2) {
                    ((float*)Cout)[(size_t)row * N + col] = val;
                } else {
                    const int bb = row >> 12, tt2 = row & (T_ - 1);
                    const int hh = col >> 7,  ee  = col & (DH_ - 1);
                    _Float16 hv;
                    if (mode == 0) hv = (_Float16)(val * qscale);
                    else           hv = (val >= 0.0f) ? (_Float16)1.0f : (_Float16)(-1.0f);
                    ((_Float16*)Cout)[(((size_t)(bb * H_ + hh)) * T_ + tt2) * DH_ + ee] = hv;
                }
            }
        }
    }
}

// ---------------- chunked linear attention ----------------
// One workgroup per (b,h). Sequential over N=64 chunks of C=64 rows.
// stT[e][d] = state[d][e] kept in fp32 LDS (prefix of sum_c k[c,d]*v[c,e]).
__global__ __launch_bounds__(256) void attn_f16(
    const _Float16* __restrict__ qb, const _Float16* __restrict__ kbuf,
    const _Float16* __restrict__ vbuf, _Float16* __restrict__ att,
    float* __restrict__ finalState)
{
    __shared__ __align__(16) _Float16 qs [CC_ * DH_];   // 16 KB
    __shared__ __align__(16) _Float16 ks [CC_ * DH_];   // 16 KB
    __shared__ __align__(16) _Float16 vs [CC_ * DH_];   // 16 KB
    __shared__ __align__(16) _Float16 ktT[DH_ * CC_];   // 16 KB  k transposed (d x c)
    __shared__ __align__(16) _Float16 vtT[DH_ * CC_];   // 16 KB  v transposed (e x c)
    __shared__ __align__(16) _Float16 sc [CC_ * CC_];   //  8 KB  masked scores
    __shared__ __align__(16) float    stT[DH_ * DH_];   // 64 KB  state^T in fp32

    const int tid  = threadIdx.x;
    const int lane = tid & 31;
    const int wave = tid >> 5;
    const int lr = lane & 15, lh = lane >> 4;
    const int bh = blockIdx.x;
    const int b = bh >> 4, h = bh & 15;

    for (int i = tid; i < DH_ * DH_; i += 256) stT[i] = 0.0f;
    __syncthreads();

    for (int n = 0; n < T_ / CC_; ++n) {
        // ---- load q/k/v chunk (contiguous 16KB each in (B,H,T,dh) layout) ----
        const size_t base = ((size_t)bh * T_ + (size_t)n * CC_) * DH_;
#pragma unroll
        for (int i = 0; i < 4; ++i) {
            const int s2 = (tid + 256 * i) << 3;    // element offset, 8 f16 per b128
            copy_b128(qb   + base + s2, qs + s2);
            copy_b128(kbuf + base + s2, ks + s2);
            copy_b128(vbuf + base + s2, vs + s2);
        }
        async_wait<0>();
        __syncthreads();
        // ---- build transposed copies for B/A operands ----
        for (int i = tid; i < CC_ * DH_; i += 256) {
            const int c = i >> 7, e = i & (DH_ - 1);
            ktT[e * CC_ + c] = ks[i];
            vtT[e * CC_ + c] = vs[i];
        }
        __syncthreads();

        // ---- phase 1: masked scores = tril(qc . kc^T), 16 tiles / 8 waves ----
        for (int tt = wave * 2; tt < wave * 2 + 2; ++tt) {
            const int ti = tt >> 2, tj = tt & 3;
            v8f a = zeroV();
            for (int kk = 0; kk < DH_; kk += 32)
                a = wmma32(load_fragA(qs, DH_, ti * 16, kk, lane),
                           load_fragB(ks, DH_, tj * 16, kk, lane), a);
#pragma unroll
            for (int r = 0; r < 8; ++r) {
                const int i2 = ti * 16 + lh * 8 + r;
                const int j2 = tj * 16 + lr;
                sc[i2 * CC_ + j2] = (_Float16)((j2 <= i2) ? a[r] : 0.0f);
            }
        }
        __syncthreads();

        // ---- phase 2: out = scores.vc + qc.prefix  (32 tiles / 8 waves) ----
        for (int idx = wave * 4; idx < wave * 4 + 4; ++idx) {
            const int ti = idx >> 3, te = idx & 7;
            v8f a = zeroV();
            for (int kk = 0; kk < CC_; kk += 32)          // intra
                a = wmma32(load_fragA(sc, CC_, ti * 16, kk, lane),
                           load_fragB(vtT, CC_, te * 16, kk, lane), a);
            for (int kk = 0; kk < DH_; kk += 32)          // cross (prefix state)
                a = wmma32(load_fragA(qs, DH_, ti * 16, kk, lane),
                           load_fragB_f32(stT, DH_, te * 16, kk, lane), a);
#pragma unroll
            for (int r = 0; r < 8; ++r) {
                const int trow = n * CC_ + ti * 16 + lh * 8 + r;
                const int e    = te * 16 + lr;
                att[((size_t)(b * T_ + trow)) * D_ + h * DH_ + e] = (_Float16)a[r];
            }
        }
        __syncthreads();

        // ---- phase 3: state^T += vc^T . kc  (64 tiles / 8 waves) ----
        for (int t2 = 0; t2 < 8; ++t2) {
            v8f cacc;
#pragma unroll
            for (int r = 0; r < 8; ++r)
                cacc[r] = stT[(wave * 16 + lh * 8 + r) * DH_ + t2 * 16 + lr];
            for (int kk = 0; kk < CC_; kk += 32)
                cacc = wmma32(load_fragA(vtT, CC_, wave * 16, kk, lane),
                              load_fragB(ktT, CC_, t2 * 16, kk, lane), cacc);
#pragma unroll
            for (int r = 0; r < 8; ++r)
                stT[(wave * 16 + lh * 8 + r) * DH_ + t2 * 16 + lr] = cacc[r];
        }
        __syncthreads();
    }

    // ---- final state: finalState[bh][i][j] = stT[j][i] ----
    for (int i = tid; i < DH_ * DH_; i += 256) {
        const int r = i >> 7, c2 = i & (DH_ - 1);
        finalState[(size_t)bh * (DH_ * DH_) + i] = stT[c2 * DH_ + r];
    }
}

// ---------------- launcher ----------------
extern "C" void kernel_launch(void* const* d_in, const int* in_sizes, int n_in,
                              void* d_out, int out_size, void* d_ws, size_t ws_size,
                              hipStream_t stream) {
    const float* x  = (const float*)d_in[0];
    const float* Wq = (const float*)d_in[1];
    const float* Wk = (const float*)d_in[2];
    const float* Wv = (const float*)d_in[3];
    const float* Wo = (const float*)d_in[4];

    const size_t NX = (size_t)BT_ * D_;   // 16,777,216
    const size_t NW = (size_t)D_ * D_;    //  4,194,304

    _Float16* ws = (_Float16*)d_ws;
    _Float16* xb   = ws;                 // A for projections (later reused as att)
    _Float16* wqb  = xb  + NX;
    _Float16* wkb  = wqb + NW;
    _Float16* wvb  = wkb + NW;
    _Float16* wob  = wvb + NW;
    _Float16* qbuf = wob + NW;           // (B,H,T,dh) f16
    _Float16* kbuf = qbuf + NX;
    _Float16* vbuf = kbuf + NX;
    _Float16* att  = xb;                 // reuse: x panel consumed before attn writes

    // 1) convert inputs to f16
    cvt_f32_f16<<<(int)(NX / 256), 256, 0, stream>>>(x,  xb,  (int)NX);
    cvt_f32_f16<<<(int)(NW / 256), 256, 0, stream>>>(Wq, wqb, (int)NW);
    cvt_f32_f16<<<(int)(NW / 256), 256, 0, stream>>>(Wk, wkb, (int)NW);
    cvt_f32_f16<<<(int)(NW / 256), 256, 0, stream>>>(Wv, wvb, (int)NW);
    cvt_f32_f16<<<(int)(NW / 256), 256, 0, stream>>>(Wo, wob, (int)NW);

    // 2) projections: q (scaled), k,v (binarized), scattered into (B,H,T,dh)
    const int grid = (BT_ / 128) * (D_ / 128);   // 1024 workgroups
    gemm_f16<<<grid, 256, 0, stream>>>(xb, wqb, (void*)qbuf, BT_, D_, D_, 0);
    gemm_f16<<<grid, 256, 0, stream>>>(xb, wkb, (void*)kbuf, BT_, D_, D_, 1);
    gemm_f16<<<grid, 256, 0, stream>>>(xb, wvb, (void*)vbuf, BT_, D_, D_, 1);

    // 3) chunked linear attention; writes att (B,T,H*dh) + final_state tail of d_out
    float* fstate = (float*)d_out + NX;
    attn_f16<<<2 * H_, 256, 0, stream>>>(qbuf, kbuf, vbuf, att, fstate);

    // 4) out = att @ Wo^T  -> fp32 into d_out
    gemm_f16<<<grid, 256, 0, stream>>>(att, wob, d_out, BT_, D_, D_, 2);
}
